// NemotronHMambaLayer_14757507629231
// MI455X (gfx1250) — matrix operand order repaired and checked by reading, not compiled, not run on hardware
//
#include <hip/hip_runtime.h>
#include <stdint.h>

// ---------------- problem constants ----------------
#define D_MODEL   1024
#define NH        32        // heads
#define PP        64        // head dim
#define NN        64        // state dim
#define GG        8         // groups
#define KCONV     4
#define D_INNER   2048      // NH*PP
#define CONV_DIM  3072      // D_INNER + 2*GG*NN
#define D_PROJ    5152      // 2*D_INNER + 2*GG*NN + NH
#define GRP_SZ    256       // D_INNER/GG
#define SEQ       2048
#define EPSF      1e-5f

typedef unsigned short bfu16;
typedef __attribute__((ext_vector_type(16))) __bf16 v16bf;
typedef __attribute__((ext_vector_type(8)))  float  v8f;

union Frag { v16bf v; uint4 q[2]; };

#if defined(__has_builtin)
#if __has_builtin(__builtin_amdgcn_global_load_async_to_lds_b32) && \
    __has_builtin(__builtin_amdgcn_s_wait_asynccnt)
#define USE_ASYNC_LDS 1
typedef __attribute__((address_space(1))) int as1_i32;   // global
typedef __attribute__((address_space(3))) int as3_i32;   // LDS
#endif
#endif

__device__ __forceinline__ bfu16 f2bf(float f) {
    uint32_t u = __float_as_uint(f);
    uint32_t r = u + 0x7FFFu + ((u >> 16) & 1u);   // round-to-nearest-even
    return (bfu16)(r >> 16);
}
__device__ __forceinline__ float siluf(float x) { return x / (1.f + __expf(-x)); }

// ---------------- 1) RMSNorm -> bf16 activations ----------------
__global__ __launch_bounds__(256)
void k_rmsnorm(const float* __restrict__ hs, const float* __restrict__ w,
               bfu16* __restrict__ xb) {
    int s = blockIdx.x, tid = threadIdx.x;
    const float* row = hs + (size_t)s * D_MODEL;
    float v[4], local = 0.f;
    #pragma unroll
    for (int i = 0; i < 4; i++) { v[i] = row[tid + i * 256]; local += v[i] * v[i]; }
    __shared__ float red[256];
    red[tid] = local; __syncthreads();
    for (int off = 128; off > 0; off >>= 1) {
        if (tid < off) red[tid] += red[tid + off];
        __syncthreads();
    }
    float scale = rsqrtf(red[0] * (1.f / D_MODEL) + EPSF);
    #pragma unroll
    for (int i = 0; i < 4; i++) {
        int d = tid + i * 256;
        xb[(size_t)s * D_MODEL + d] = f2bf(v[i] * scale * w[d]);
    }
}

// ---------------- 2) fp32 [K][N] -> bf16 [N][K] weight transpose ----------------
__global__ __launch_bounds__(256)
void k_transpose_bf16(const float* __restrict__ W, bfu16* __restrict__ Wt, int K, int N) {
    int k = blockIdx.x * 256 + threadIdx.x;
    int n = blockIdx.y;
    Wt[(size_t)n * K + k] = f2bf(W[(size_t)k * N + n]);
}

// ---------------- 3) WMMA bf16 GEMM: C[M][N] = A[M][K] * Wt[N][K]^T (+resid) ----------------
// wave tile: 2x4 WMMA tiles (32 rows x 64 cols); 8 waves = 4x2 -> 128x128 per workgroup.
// K-loop unrolled x2 with double-buffered fragments so loads for k+32 issue before
// the WMMAs consuming k (load/matrix-pipe overlap).
struct Frags { Frag a0, a1, b0, b1, b2, b3; };

__device__ __forceinline__
void load_frags(Frags& f,
                const bfu16* __restrict__ aR0, const bfu16* __restrict__ aR1,
                const bfu16* __restrict__ bR0, const bfu16* __restrict__ bR1,
                const bfu16* __restrict__ bR2, const bfu16* __restrict__ bR3,
                int k0, int hi) {
    // A 16x32 bf16 fragment: lanes 0-15 K=0..15, lanes 16-31 K=16..31 (row = lane&15)
    f.a0.q[0] = *(const uint4*)(aR0 + k0 + hi * 16);
    f.a0.q[1] = *(const uint4*)(aR0 + k0 + hi * 16 + 8);
    f.a1.q[0] = *(const uint4*)(aR1 + k0 + hi * 16);
    f.a1.q[1] = *(const uint4*)(aR1 + k0 + hi * 16 + 8);
    // B 32x16 bf16 fragment: lanes 0-15 K=0..15, lanes 16-31 K=16..31 (col = lane&15)
    f.b0.q[0] = *(const uint4*)(bR0 + k0 + hi * 16);
    f.b0.q[1] = *(const uint4*)(bR0 + k0 + hi * 16 + 8);
    f.b1.q[0] = *(const uint4*)(bR1 + k0 + hi * 16);
    f.b1.q[1] = *(const uint4*)(bR1 + k0 + hi * 16 + 8);
    f.b2.q[0] = *(const uint4*)(bR2 + k0 + hi * 16);
    f.b2.q[1] = *(const uint4*)(bR2 + k0 + hi * 16 + 8);
    f.b3.q[0] = *(const uint4*)(bR3 + k0 + hi * 16);
    f.b3.q[1] = *(const uint4*)(bR3 + k0 + hi * 16 + 8);
}

__device__ __forceinline__
void mma_frags(const Frags& f, v8f acc[2][4]) {
    acc[0][0] = __builtin_amdgcn_wmma_f32_16x16x32_bf16(false, f.a0.v, false, f.b0.v, (short)0, acc[0][0], false, false);
    acc[0][1] = __builtin_amdgcn_wmma_f32_16x16x32_bf16(false, f.a0.v, false, f.b1.v, (short)0, acc[0][1], false, false);
    acc[0][2] = __builtin_amdgcn_wmma_f32_16x16x32_bf16(false, f.a0.v, false, f.b2.v, (short)0, acc[0][2], false, false);
    acc[0][3] = __builtin_amdgcn_wmma_f32_16x16x32_bf16(false, f.a0.v, false, f.b3.v, (short)0, acc[0][3], false, false);
    acc[1][0] = __builtin_amdgcn_wmma_f32_16x16x32_bf16(false, f.a1.v, false, f.b0.v, (short)0, acc[1][0], false, false);
    acc[1][1] = __builtin_amdgcn_wmma_f32_16x16x32_bf16(false, f.a1.v, false, f.b1.v, (short)0, acc[1][1], false, false);
    acc[1][2] = __builtin_amdgcn_wmma_f32_16x16x32_bf16(false, f.a1.v, false, f.b2.v, (short)0, acc[1][2], false, false);
    acc[1][3] = __builtin_amdgcn_wmma_f32_16x16x32_bf16(false, f.a1.v, false, f.b3.v, (short)0, acc[1][3], false, false);
}

template<bool ADD_RESID>
__device__ __forceinline__
void store_tile(float* __restrict__ C, const float* __restrict__ resid,
                v8f d, int rowTile, int colTile, int lane, int N) {
    int m = lane & 15, hi = lane >> 4;
    int col = colTile + m;
    if (col >= N) return;
    #pragma unroll
    for (int r = 0; r < 8; r++) {
        size_t idx = (size_t)(rowTile + hi * 8 + r) * N + col;
        float v = d[r];
        if (ADD_RESID) v += resid[idx];
        C[idx] = v;
    }
}

template<bool ADD_RESID>
__global__ __launch_bounds__(256)
void k_gemm_wmma(const bfu16* __restrict__ A, const bfu16* __restrict__ Wt,
                 float* __restrict__ C, const float* __restrict__ resid,
                 int M, int N, int K) {
    const int lane = threadIdx.x;
    const int wave = threadIdx.y;
    const int waveRow = wave & 3, waveCol = wave >> 2;
    const int rowBase = blockIdx.y * 128 + waveRow * 32;
    const int colBase = blockIdx.x * 128 + waveCol * 64;
    const int m = lane & 15, hi = lane >> 4;

    v8f zero = {};
    v8f acc[2][4];
    #pragma unroll
    for (int i = 0; i < 2; i++)
        #pragma unroll
        for (int j = 0; j < 4; j++) acc[i][j] = zero;

    const bfu16* aR0 = A + (size_t)(rowBase + m) * K;
    const bfu16* aR1 = A + (size_t)(rowBase + 16 + m) * K;
    int n0 = colBase + m;      if (n0 >= N) n0 = N - 1;   // clamp ragged N (stores guarded)
    int n1 = colBase + 16 + m; if (n1 >= N) n1 = N - 1;
    int n2 = colBase + 32 + m; if (n2 >= N) n2 = N - 1;
    int n3 = colBase + 48 + m; if (n3 >= N) n3 = N - 1;
    const bfu16* bR0 = Wt + (size_t)n0 * K;
    const bfu16* bR1 = Wt + (size_t)n1 * K;
    const bfu16* bR2 = Wt + (size_t)n2 * K;
    const bfu16* bR3 = Wt + (size_t)n3 * K;

    Frags f0, f1;
    load_frags(f0, aR0, aR1, bR0, bR1, bR2, bR3, 0, hi);
    for (int k0 = 0; k0 < K; k0 += 64) {               // K is a multiple of 64
        load_frags(f1, aR0, aR1, bR0, bR1, bR2, bR3, k0 + 32, hi);
        if (k0 + 128 < K) {                            // global_prefetch_b8 two slices ahead
            __builtin_prefetch(aR0 + k0 + 128, 0, 1);
            __builtin_prefetch(bR0 + k0 + 128, 0, 1);
        }
        mma_frags(f0, acc);
        if (k0 + 64 < K)
            load_frags(f0, aR0, aR1, bR0, bR1, bR2, bR3, k0 + 64, hi);
        mma_frags(f1, acc);
    }

    #pragma unroll
    for (int i = 0; i < 2; i++)
        #pragma unroll
        for (int j = 0; j < 4; j++)
            store_tile<ADD_RESID>(C, resid, acc[i][j], rowBase + i * 16, colBase + j * 16, lane, N);
}

// ---------------- 4) depthwise causal conv (K=4) + SiLU ----------------
__global__ __launch_bounds__(256)
void k_conv_silu(const float* __restrict__ proj, const float* __restrict__ cw,
                 const float* __restrict__ cb, float* __restrict__ xact) {
    int c = blockIdx.x * 256 + threadIdx.x;
    int s = blockIdx.y;
    float acc = cb[c];
    #pragma unroll
    for (int k = 0; k < KCONV; k++) {
        int t = s + k - (KCONV - 1);
        if (t >= 0) acc += cw[c * KCONV + k] * proj[(size_t)t * D_PROJ + D_INNER + c];
    }
    xact[(size_t)s * CONV_DIM + c] = siluf(acc);
}

// ---------------- 5) dt = softplus(dt_raw + bias); dA = exp(dt * -exp(A_log)) ----------------
__global__ __launch_bounds__(256)
void k_dt(const float* __restrict__ proj, const float* __restrict__ dt_bias,
          const float* __restrict__ A_log, float* __restrict__ dt, float* __restrict__ dA) {
    int idx = blockIdx.x * 256 + threadIdx.x;     // SEQ*NH
    int h = idx & (NH - 1), s = idx >> 5;
    float x = proj[(size_t)s * D_PROJ + (D_INNER + CONV_DIM) + h] + dt_bias[h];
    float sp = (x > 20.f) ? x : log1pf(__expf(x));
    float Av = -__expf(A_log[h]);
    dt[idx] = sp;
    dA[idx] = __expf(sp * Av);
}

// ---------------- 6) selective scan: one workgroup per head ----------------
// state h[P=64][N=64] in registers: thread owns p = tid>>2, n-quarter = tid&3 (16 floats)
__global__ __launch_bounds__(256)
void k_scan(const float* __restrict__ xact, const float* __restrict__ dt,
            const float* __restrict__ dA, float* __restrict__ y) {
    int head = blockIdx.x, tid = threadIdx.x;
    int p = tid >> 2, nq = tid & 3;
    int g = head >> 2;                             // H/G = 4 heads per group
    int xbase = head * PP;
    int Bbase = D_INNER + g * NN;
    int Cbase = D_INNER + GG * NN + g * NN;
    __shared__ float sB[NN], sC[NN], sx[PP];
    float h[16];
    #pragma unroll
    for (int u = 0; u < 16; u++) h[u] = 0.f;

    // per-lane global source / LDS destination for the staged row pieces
    const int li  = (tid < 64) ? tid : (tid < 128 ? tid - 64 : tid - 128);
    const int src = (tid < 64) ? (Bbase + li) : (tid < 128 ? (Cbase + li) : (xbase + li));
    float* dst    = (tid < 64) ? &sB[li] : (tid < 128 ? &sC[li] : &sx[li]);

    for (int t = 0; t < SEQ; t++) {
        const float* row = xact + (size_t)t * CONV_DIM;
        if (tid < 192) {
#if defined(USE_ASYNC_LDS)
            // GLOBAL_LOAD_ASYNC_TO_LDS_B32 (ASYNCcnt-tracked): global AS1 src, LDS AS3 dst
            __builtin_amdgcn_global_load_async_to_lds_b32(
                (as1_i32*)(row + src),
                (as3_i32*)dst, 0, 0);
#else
            *dst = row[src];
#endif
        }
#if defined(USE_ASYNC_LDS)
        __builtin_amdgcn_s_wait_asynccnt(0);
#endif
        __syncthreads();
        float dAv = dA[t * NH + head];
        float dtv = dt[t * NH + head];
        float coeff = dtv * sx[p];
        float partial = 0.f;
        #pragma unroll
        for (int u = 0; u < 16; u++) {
            int n = nq * 16 + u;
            h[u] = dAv * h[u] + coeff * sB[n];
            partial += h[u] * sC[n];
        }
        // reduce the 4 n-quarters sharing the same p (lanes xor 1, xor 2)
        partial += __shfl_xor(partial, 1, 32);
        partial += __shfl_xor(partial, 2, 32);
        if (nq == 0) y[(size_t)t * D_INNER + head * PP + p] = partial;
        __syncthreads();
    }
}

// ---------------- 7) y += D*xs ; g = y*silu(z) ; grouped RMS-scale ; -> bf16 ----------------
__global__ __launch_bounds__(256)
void k_gate(const float* __restrict__ y, const float* __restrict__ xact,
            const float* __restrict__ proj, const float* __restrict__ Dvec,
            const float* __restrict__ gw, bfu16* __restrict__ gb) {
    int s = blockIdx.y, grp = blockIdx.x, tid = threadIdx.x;
    int c = grp * GRP_SZ + tid;
    int head = c >> 6;
    float xs = xact[(size_t)s * CONV_DIM + c];
    float yv = y[(size_t)s * D_INNER + c] + Dvec[head] * xs;
    float z  = proj[(size_t)s * D_PROJ + c];
    float gval = yv * siluf(z);
    __shared__ float red[256];
    red[tid] = gval * gval; __syncthreads();
    for (int off = 128; off > 0; off >>= 1) {
        if (tid < off) red[tid] += red[tid + off];
        __syncthreads();
    }
    float scale = rsqrtf(red[0] * (1.f / GRP_SZ) + EPSF);
    gb[(size_t)s * D_INNER + c] = f2bf(gval * scale * gw[c]);
}

// ---------------- launcher ----------------
extern "C" void kernel_launch(void* const* d_in, const int* in_sizes, int n_in,
                              void* d_out, int out_size, void* d_ws, size_t ws_size,
                              hipStream_t stream) {
    const float* hs          = (const float*)d_in[0];
    const float* norm_w      = (const float*)d_in[1];
    const float* in_proj_w   = (const float*)d_in[2];
    const float* conv_w      = (const float*)d_in[3];
    const float* conv_b      = (const float*)d_in[4];
    const float* dt_bias     = (const float*)d_in[5];
    const float* A_log       = (const float*)d_in[6];
    const float* Dvec        = (const float*)d_in[7];
    const float* gate_norm_w = (const float*)d_in[8];
    const float* out_proj_w  = (const float*)d_in[9];
    float* out = (float*)d_out;
    (void)in_sizes; (void)n_in; (void)out_size; (void)ws_size;

    char* ws = (char*)d_ws;
    size_t off = 0;
    auto alloc = [&](size_t bytes) -> void* {
        void* p = ws + off;
        off = (off + bytes + 255) & ~(size_t)255;
        return p;
    };
    bfu16* xb   = (bfu16*)alloc((size_t)SEQ * D_MODEL * sizeof(bfu16));
    bfu16* Wt1  = (bfu16*)alloc((size_t)D_PROJ * D_MODEL * sizeof(bfu16));
    bfu16* Wt2  = (bfu16*)alloc((size_t)D_MODEL * D_INNER * sizeof(bfu16));
    float* proj = (float*)alloc((size_t)SEQ * D_PROJ * sizeof(float));
    float* xact = (float*)alloc((size_t)SEQ * CONV_DIM * sizeof(float));
    float* dtb  = (float*)alloc((size_t)SEQ * NH * sizeof(float));
    float* dAb  = (float*)alloc((size_t)SEQ * NH * sizeof(float));
    float* ybuf = (float*)alloc((size_t)SEQ * D_INNER * sizeof(float));
    bfu16* gb   = (bfu16*)alloc((size_t)SEQ * D_INNER * sizeof(bfu16));

    // 1) RMSNorm + bf16 convert
    k_rmsnorm<<<SEQ, 256, 0, stream>>>(hs, norm_w, xb);
    // 2) weight transposes to bf16 [N][K]
    k_transpose_bf16<<<dim3(D_MODEL / 256, D_PROJ), 256, 0, stream>>>(in_proj_w, Wt1, D_MODEL, D_PROJ);
    k_transpose_bf16<<<dim3(D_INNER / 256, D_MODEL), 256, 0, stream>>>(out_proj_w, Wt2, D_INNER, D_MODEL);
    // 3) in_proj GEMM (WMMA bf16, f32 accum): proj = xb @ in_proj_w
    k_gemm_wmma<false><<<dim3((D_PROJ + 127) / 128, SEQ / 128), dim3(32, 8), 0, stream>>>(
        xb, Wt1, proj, nullptr, SEQ, D_PROJ, D_MODEL);
    // 4) causal depthwise conv + SiLU
    k_conv_silu<<<dim3(CONV_DIM / 256, SEQ), 256, 0, stream>>>(proj, conv_w, conv_b, xact);
    // 5) dt / dA
    k_dt<<<(SEQ * NH) / 256, 256, 0, stream>>>(proj, dt_bias, A_log, dtb, dAb);
    // 6) selective scan (one WG per head)
    k_scan<<<NH, 256, 0, stream>>>(xact, dtb, dAb, ybuf);
    // 7) gating + grouped RMS scale -> bf16
    k_gate<<<dim3(GG, SEQ), 256, 0, stream>>>(ybuf, xact, proj, Dvec, gate_norm_w, gb);
    // 8) out_proj GEMM + residual: out = hs + gb @ out_proj_w
    k_gemm_wmma<true><<<dim3(D_MODEL / 128, SEQ / 128), dim3(32, 8), 0, stream>>>(
        gb, Wt2, out, hs, SEQ, D_MODEL, D_INNER);
}
